// SelfAttention_38001870635635
// MI455X (gfx1250) — matrix operand order repaired and checked by reading, not compiled, not run on hardware
//
#include <hip/hip_runtime.h>
#include <hip/hip_bf16.h>

// ---------------------------------------------------------------------------
// Self-attention for MI455X (gfx1250): bf16 WMMA pipeline.
//   B=8, S=2048, D=768, DIM_K=DIM_V=768.  NORM = 1/sqrt(768).
// ---------------------------------------------------------------------------

typedef __bf16 bf16_t;
typedef __attribute__((ext_vector_type(16))) __bf16 v16bf;
typedef __attribute__((ext_vector_type(8)))  __bf16 v8bf;
typedef __attribute__((ext_vector_type(8)))  float  v8f;

#define BATCH 8
#define SEQ   2048
#define DIM   768
#define NORM  0.036084391824351615f   // 1/sqrt(768)

__device__ __forceinline__ v16bf frag_cat(v8bf lo, v8bf hi) {
    union { v16bf v; v8bf h[2]; } u;
    u.h[0] = lo; u.h[1] = hi;
    return u.v;
}

// A-matrix 16x32 bf16 fragment from a row-major [16 x ld] bf16 tile.
// Layout: lane&15 = M; lanes 16..31 have k-base 16; VGPR j holds
// k=kb+2j,kb+2j+1 (matches load_b_frag).  Two 16B chunks per lane.
__device__ __forceinline__ v16bf load_a_frag(const bf16_t* base, int ld, int k0, int lane) {
    int m  = lane & 15;
    int kb = (lane >> 4) << 4;           // 0 or 16
    const bf16_t* p = base + (size_t)m * ld + k0 + kb;
    v8bf lo = *(const v8bf*)(p);
    v8bf hi = *(const v8bf*)(p + 8);
    return frag_cat(lo, hi);
}

// B-matrix 32x16 bf16 fragment where B[k][n] = M[n][k] for row-major M with
// stride ld.  Layout used here: lane&15 = N; lanes 16..31 have k-base 16; VGPR j
// holds k=kb+2j,kb+2j+1.  32 contiguous bytes per lane (32B aligned).
__device__ __forceinline__ v16bf load_b_frag(const bf16_t* base, int ld, int k0, int lane) {
    int n  = lane & 15;
    int kb = (lane >> 4) << 4;           // 0 or 16
    return *(const v16bf*)(base + (size_t)n * ld + k0 + kb);
}

__device__ __forceinline__ v8f wmma_bf16(v16bf a, v16bf b, v8f c) {
    return __builtin_amdgcn_wmma_f32_16x16x32_bf16(
        /*neg_a=*/false, a, /*neg_b=*/false, b,
        /*c_mod=*/(short)0, c, /*reuse_a=*/false, /*reuse_b=*/false);
}

// ---------------------------------------------------------------------------
// Kernel 1: convert & transpose weights  WT[mat][n][d] = (bf16) W[mat][d][n]
// ---------------------------------------------------------------------------
__global__ __launch_bounds__(256) void wtr_kernel(
    const float* __restrict__ Wq, const float* __restrict__ Wk,
    const float* __restrict__ Wv, bf16_t* __restrict__ WT) {
    int idx = blockIdx.x * 256 + threadIdx.x;
    if (idx >= 3 * DIM * DIM) return;
    int mat = idx / (DIM * DIM);
    int rem = idx - mat * DIM * DIM;
    int d = rem / DIM, n = rem - (rem / DIM) * DIM;
    const float* W = (mat == 0) ? Wq : (mat == 1) ? Wk : Wv;
    WT[(size_t)mat * DIM * DIM + (size_t)n * DIM + d] = (bf16_t)W[(size_t)d * DIM + n];
}

// ---------------------------------------------------------------------------
// Kernel 2: QKV projection.  One block = one 16-row tile of x.
//   Q,K stored row-major bf16 [B*S][768]; V stored transposed VT[B][768][S].
// ---------------------------------------------------------------------------
__global__ __launch_bounds__(256) void qkv_kernel(
    const float*  __restrict__ x,
    const bf16_t* __restrict__ WT,      // [3][768][768] (Q,K,V)
    const float*  __restrict__ bq, const float* __restrict__ bk,
    const float*  __restrict__ bv,
    bf16_t* __restrict__ Qb, bf16_t* __restrict__ Kb,
    bf16_t* __restrict__ VT) {
    __shared__ __align__(32) bf16_t xt[16 * DIM];   // 24 KB

    int m0 = blockIdx.x * 16;                       // global row in [0, B*S)
    for (int e = threadIdx.x; e < 16 * DIM; e += 256)
        xt[e] = (bf16_t)x[(size_t)m0 * DIM + e];
    __syncthreads();

    int wave = threadIdx.x >> 5, lane = threadIdx.x & 31;
    int n = lane & 15, hi = lane >> 4;

    for (int t = wave; t < 3 * (DIM / 16); t += 8) {   // 144 tiles
        int mat = t / 48;
        int n0  = (t - mat * 48) * 16;
        const bf16_t* wt = WT + (size_t)mat * DIM * DIM + (size_t)n0 * DIM;
        v8f c = {};
        for (int k0 = 0; k0 < DIM; k0 += 32) {
            v16bf a = load_a_frag(xt, DIM, k0, lane);
            v16bf b = load_b_frag(wt, DIM, k0, lane);
            c = wmma_bf16(a, b, c);
        }
        const float* bp = (mat == 0) ? bq : (mat == 1) ? bk : bv;
        float bn = bp[n0 + n];
        if (mat < 2) {
            bf16_t* dst = (mat == 0) ? Qb : Kb;
            for (int r = 0; r < 8; ++r) {
                int row = r + 8 * hi;
                dst[(size_t)(m0 + row) * DIM + n0 + n] = (bf16_t)(c[r] + bn);
            }
        } else {
            int bidx = m0 / SEQ, s0 = m0 - bidx * SEQ;
            bf16_t* dst = VT + ((size_t)bidx * DIM + (n0 + n)) * SEQ;
            for (int r = 0; r < 8; ++r)
                dst[s0 + 8 * hi + r] = (bf16_t)(c[r] + bn);
        }
    }
}

// ---------------------------------------------------------------------------
// Kernel 3: attention.  One block = (batch b, 16 query rows).  8 waves.
//   Phase 1: S = Q K^T * mask  -> bf16 logits in LDS (16 x 2048)
//   Phase 2: exact softmax in place, folding * NORM
//   Phase 3: out = P @ V  via  VT  (B fragments contiguous)
// ---------------------------------------------------------------------------
__global__ __launch_bounds__(256) void attn_kernel(
    const bf16_t* __restrict__ Qb,   // [B*S][768]
    const bf16_t* __restrict__ Kb,   // [B*S][768]
    const bf16_t* __restrict__ VT,   // [B][768][S]
    const int*    __restrict__ mask, // [B][S]
    float*        __restrict__ out)  // [B][S][768]
{
    __shared__ __align__(32) bf16_t qs[16 * DIM];   // 24 KB
    __shared__ __align__(32) bf16_t ss[16 * SEQ];   // 64 KB
    __shared__ float red[16 * 16];
    __shared__ float mkv[16];

    int b  = blockIdx.y;
    int s0 = blockIdx.x * 16;

    const bf16_t* Qrow = Qb + ((size_t)b * SEQ + s0) * DIM;
    for (int e = threadIdx.x; e < 16 * DIM; e += 256) qs[e] = Qrow[e];
    if (threadIdx.x < 16)
        mkv[threadIdx.x] = (float)mask[b * SEQ + s0 + threadIdx.x];
    __syncthreads();

    int wave = threadIdx.x >> 5, lane = threadIdx.x & 31;
    int n = lane & 15, hi = lane >> 4;

    // ---- Phase 1: masked raw scores -> ss (bf16) ----
    const bf16_t* Kbase = Kb + (size_t)b * SEQ * DIM;
    for (int tt = wave; tt < SEQ / 16; tt += 8) {     // 128 tiles / 8 waves
        int t0 = tt * 16;
        const bf16_t* kt = Kbase + (size_t)t0 * DIM;
        v8f c = {};
        for (int k0 = 0; k0 < DIM; k0 += 32) {
            v16bf a = load_a_frag(qs, DIM, k0, lane);
            v16bf bb = load_b_frag(kt, DIM, k0, lane);
            c = wmma_bf16(a, bb, c);
        }
        for (int r = 0; r < 8; ++r) {
            int row = r + 8 * hi;
            ss[(size_t)row * SEQ + t0 + n] = (bf16_t)(c[r] * mkv[row]);
        }
    }
    __syncthreads();

    // ---- Phase 2: softmax per row (16 threads/row, 128 elems each) ----
    {
        int row = threadIdx.x >> 4;       // 0..15
        int i   = threadIdx.x & 15;
        bf16_t* sr = ss + (size_t)row * SEQ;
        int t0 = i * 128;

        float m = -3.0e38f;
        for (int t = t0; t < t0 + 128; ++t) m = fmaxf(m, (float)sr[t]);
        red[row * 16 + i] = m;
        __syncthreads();
        float rm = red[row * 16];
        for (int j = 1; j < 16; ++j) rm = fmaxf(rm, red[row * 16 + j]);

        float sum = 0.f;
        for (int t = t0; t < t0 + 128; ++t) sum += __expf((float)sr[t] - rm);
        __syncthreads();                  // everyone done reading maxes
        red[row * 16 + i] = sum;
        __syncthreads();
        float rs = 0.f;
        for (int j = 0; j < 16; ++j) rs += red[row * 16 + j];

        float scale = NORM / rs;          // post-softmax 1/sqrt(dk) folded in
        for (int t = t0; t < t0 + 128; ++t)
            sr[t] = (bf16_t)(__expf((float)sr[t] - rm) * scale);  // in place
    }
    __syncthreads();

    // ---- Phase 3: out = P @ V  (P: 16 x 2048 bf16 in LDS, V via VT) ----
    const bf16_t* Vb = VT + (size_t)b * DIM * SEQ;
    float* orow = out + ((size_t)b * SEQ + s0) * DIM;

    v8f acc[6];
    for (int j = 0; j < 6; ++j) acc[j] = (v8f){};
    int nbase = wave * 6 * 16;                         // 6 n-tiles per wave
    for (int k0 = 0; k0 < SEQ; k0 += 32) {
        v16bf a = load_a_frag(ss, SEQ, k0, lane);
        for (int j = 0; j < 6; ++j) {
            const bf16_t* vt = Vb + (size_t)(nbase + j * 16) * SEQ;
            v16bf bb = load_b_frag(vt, SEQ, k0, lane);
            acc[j] = wmma_bf16(a, bb, acc[j]);
        }
    }
    for (int j = 0; j < 6; ++j) {
        int n0 = nbase + j * 16;
        for (int r = 0; r < 8; ++r) {
            int row = r + 8 * hi;
            orow[(size_t)row * DIM + n0 + n] = acc[j][r];
        }
    }
}

// ---------------------------------------------------------------------------
// Launcher.  Workspace layout (bytes, 256-aligned):
//   WT : 3*768*768*2      =  3,538,944   @ 0
//   Qb : 16384*768*2      = 25,165,824   @ 3,538,944
//   Kb : 25,165,824                      @ 28,704,768
//   VT : 8*768*2048*2     = 25,165,824   @ 53,870,592
//   total ~ 79 MB
// ---------------------------------------------------------------------------
extern "C" void kernel_launch(void* const* d_in, const int* in_sizes, int n_in,
                              void* d_out, int out_size, void* d_ws, size_t ws_size,
                              hipStream_t stream) {
    const float* x    = (const float*)d_in[0];
    const int*   mask = (const int*)d_in[1];
    const float* Wq   = (const float*)d_in[2];
    const float* bq   = (const float*)d_in[3];
    const float* Wk   = (const float*)d_in[4];
    const float* bk   = (const float*)d_in[5];
    const float* Wv   = (const float*)d_in[6];
    const float* bv   = (const float*)d_in[7];
    float* out = (float*)d_out;

    char* ws = (char*)d_ws;
    bf16_t* WT = (bf16_t*)(ws + 0);
    bf16_t* Qb = (bf16_t*)(ws + 3538944);
    bf16_t* Kb = (bf16_t*)(ws + 28704768);
    bf16_t* VT = (bf16_t*)(ws + 53870592);

    // 1) weight transpose + bf16 convert
    {
        int total = 3 * DIM * DIM;
        wtr_kernel<<<(total + 255) / 256, 256, 0, stream>>>(Wq, Wk, Wv, WT);
    }
    // 2) QKV projection: one block per 16 rows of [B*S, D]
    qkv_kernel<<<(BATCH * SEQ) / 16, 256, 0, stream>>>(x, WT, bq, bk, bv, Qb, Kb, VT);
    // 3) attention: grid (S/16, B)
    attn_kernel<<<dim3(SEQ / 16, BATCH), 256, 0, stream>>>(Qb, Kb, VT, mask, out);
}
